// Update_67834713473121
// MI455X (gfx1250) — matrix operand order repaired and checked by reading, not compiled, not run on hardware
//
#include <hip/hip_runtime.h>
#include <hip/hip_bf16.h>
#include <cstdint>

typedef __attribute__((ext_vector_type(16))) _Float16 v16h;
typedef __attribute__((ext_vector_type(8)))  _Float16 v8h;
typedef __attribute__((ext_vector_type(8)))  float    v8f;

#define BN   4
#define HD   32
#define HH   256
#define WW   320
#define CPAD 192      // 161 channels of hx padded to 6*32
#define NS   256

#define XP_H 258      // pad-1 halo for 3x3
#define XP_W 322
#define HP_H 260      // pad-2 halo for 3x3 dil2
#define HP_W 324

static constexpr size_t XPACK_ELEMS = (size_t)BN*XP_H*XP_W*CPAD;
static constexpr size_t HPACK_ELEMS = (size_t)BN*HP_H*HP_W*HD;
static constexpr size_t PIX         = (size_t)BN*HH*WW;          // 327680
static constexpr int    NWAVE16     = BN*HH*(WW/16);             // 20480 16-px strips
static constexpr int    NBLK16      = NWAVE16/8;                 // 2560
static constexpr int    NWAVE32     = BN*HH*(WW/32);             // 10240 32-px strips
static constexpr int    NBLK32      = NWAVE32/8;                 // 1280

// ---- workspace layout (bytes) ----
static constexpr size_t OFF_XPACK = 0;
static constexpr size_t SZ_XPACK  = XPACK_ELEMS*2;
static constexpr size_t OFF_HPACK = OFF_XPACK + SZ_XPACK;
static constexpr size_t SZ_HPACK  = HPACK_ELEMS*2;
static constexpr size_t OFF_Z     = OFF_HPACK + SZ_HPACK;
static constexpr size_t SZ_ZR     = (size_t)BN*HD*HH*WW*4;
static constexpr size_t OFF_R     = OFF_Z + SZ_ZR;
static constexpr size_t OFF_T1    = OFF_R + SZ_ZR;
static constexpr size_t SZ_T1     = PIX*32*2;
static constexpr size_t OFF_C1    = OFF_T1 + SZ_T1;
static constexpr size_t OFF_T2    = OFF_C1 + SZ_T1;
static constexpr size_t SZ_T2     = PIX*64*2;
static constexpr size_t OFF_WZ    = OFF_T2 + SZ_T2;
static constexpr size_t SZ_WGRU   = (size_t)54*2*512*2;   // [54 kc][2 nt][32 lane][16]
static constexpr size_t OFF_WR    = OFF_WZ + SZ_WGRU;
static constexpr size_t OFF_WQ    = OFF_WR + SZ_WGRU;
static constexpr size_t OFF_WD1   = OFF_WQ + SZ_WGRU;
static constexpr size_t SZ_W33H   = (size_t)9*2*512*2;
static constexpr size_t OFF_WC1   = OFF_WD1 + SZ_W33H;
static constexpr size_t OFF_WD2   = OFF_WC1 + SZ_W33H;
static constexpr size_t SZ_WD2    = (size_t)4*512*2;
static constexpr size_t OFF_WD3   = OFF_WD2 + SZ_WD2;

// ---------------- device helpers ----------------
__device__ __forceinline__ v16h load_a16(const _Float16* p) {
    // 16-bit A-matrix layout: this lane's 16 halfs = [K=c..c+7] and [K=c+16..c+23]
    v8h lo = *(const v8h*)p;
    v8h hi = *(const v8h*)(p + 16);
    v16h a;
#pragma unroll
    for (int i = 0; i < 8; ++i) { a[i] = lo[i]; a[i+8] = hi[i]; }
    return a;
}

__device__ __forceinline__ v8f wmma16(v16h a, v16h b, v8f c) {
    return __builtin_amdgcn_wmma_f32_16x16x32_f16(false, a, false, b, (short)0, c, false, false);
}

__device__ __forceinline__ float sigm(float x) { return 1.f/(1.f + __expf(-x)); }

__device__ __forceinline__ void strip_decode16(int wave, int& b, int& y, int& xs) {
    b = wave / (HH*(WW/16));
    int rem = wave % (HH*(WW/16));
    y = rem / (WW/16);
    xs = rem % (WW/16);
}

__device__ __forceinline__ void strip_decode32(int wave, int& b, int& y, int& xs) {
    b = wave / (HH*(WW/32));
    int rem = wave % (HH*(WW/32));
    y = rem / (WW/32);
    xs = rem % (WW/32);
}

// ---------------- packing kernels ----------------
// dst layout: [9*cchunks kc][Cout/16 nt][32 lane][16 e]; dense f16 B layout:
// h = lane>>4: e<8 -> K local = 8h+e, e>=8 -> K local = 16+8h+(e-8); N = nt*16 + (lane&15)
__global__ void pack_w33_kernel(const float* __restrict__ src, _Float16* __restrict__ dst,
                                int Cout, int Cin, int cchunks) {
    int i = blockIdx.x*blockDim.x + threadIdx.x;
    int NT = Cout >> 4;
    int total = 9*cchunks*NT*512;
    if (i >= total) return;
    int e    = i & 15;
    int lane = (i >> 4) & 31;
    int nt   = (i >> 9) % NT;
    int kc   = i / (NT << 9);
    int t    = kc / cchunks;
    int cc   = kc - t*cchunks;
    int n    = nt*16 + (lane & 15);
    int ci   = cc*32 + ((e & 8) << 1) + ((lane >> 4) << 3) + (e & 7);
    float v = 0.f;
    if (ci < Cin) v = src[(size_t)(n*Cin + ci)*9 + t];
    dst[i] = (_Float16)v;
}

__global__ void pack_w11_kernel(const float* __restrict__ src, _Float16* __restrict__ dst,
                                int Cout, int Cin) {
    int i = blockIdx.x*blockDim.x + threadIdx.x;
    int NT = Cout >> 4;
    int total = (Cin >> 5)*NT*512;
    if (i >= total) return;
    int e    = i & 15;
    int lane = (i >> 4) & 31;
    int nt   = (i >> 9) % NT;
    int kc   = i / (NT << 9);
    int n    = nt*16 + (lane & 15);
    int ci   = kc*32 + ((e & 8) << 1) + ((lane >> 4) << 3) + (e & 7);
    dst[i] = (_Float16)src[n*Cin + ci];
}

// xpack [B][258][322][192] f16, halo + channel padding zeroed.
// channels: 0..31 hidden, 32 depth, 33..160 corr, 161..191 zero
__global__ void pack_x_kernel(const float* __restrict__ hidden, const float* __restrict__ depth,
                              const float* __restrict__ corr, _Float16* __restrict__ xp) {
    size_t i = (size_t)blockIdx.x*blockDim.x + threadIdx.x;
    if (i >= XPACK_ELEMS) return;
    int c  = (int)(i % CPAD); size_t r = i / CPAD;
    int x  = (int)(r % XP_W) - 1; r /= XP_W;
    int y  = (int)(r % XP_H) - 1;
    int b  = (int)(r / XP_H);
    float v = 0.f;
    if (x >= 0 && x < WW && y >= 0 && y < HH && c < 161) {
        if (c < 32)       v = hidden[(((size_t)(b*HD + c))*HH + y)*WW + x];
        else if (c == 32) v = depth[((size_t)b*HH + y)*WW + x];
        else              v = corr[(((size_t)(b*128 + (c-33)))*HH + y)*WW + x];
    }
    xp[i] = (_Float16)v;
}

__global__ void zero32_kernel(uint32_t* __restrict__ p, size_t n) {
    size_t i = (size_t)blockIdx.x*blockDim.x + threadIdx.x;
    if (i < n) p[i] = 0u;
}

// overwrite xpack channels 0..31 with r*hidden (interior only)
__global__ void rhpack_kernel(const float* __restrict__ r, const float* __restrict__ hidden,
                              _Float16* __restrict__ xp) {
    size_t i = (size_t)blockIdx.x*blockDim.x + threadIdx.x;
    if (i >= PIX*HD) return;
    int c = (int)(i & 31); size_t p = i >> 5;
    int x = (int)(p % WW); p /= WW;
    int y = (int)(p % HH); int b = (int)(p / HH);
    size_t idx = (((size_t)(b*HD + c))*HH + y)*WW + x;
    xp[(((size_t)b*XP_H + (y+1))*XP_W + (x+1))*CPAD + c] = (_Float16)(r[idx]*hidden[idx]);
}

// ---------------- ConvGRU z,r : implicit GEMM, M=32 (two A tiles share B frags) ----------
__global__ __launch_bounds__(256) void gru_zr_kernel(
    const _Float16* __restrict__ xp, const _Float16* __restrict__ wz,
    const _Float16* __restrict__ wr, const float* __restrict__ bz,
    const float* __restrict__ br, float* __restrict__ zout, float* __restrict__ rout) {
    int wave = (int)((blockIdx.x*blockDim.x + threadIdx.x) >> 5);
    if (wave >= NWAVE32) return;
    int lane = threadIdx.x & 31, m = lane & 15, hi = lane >> 4;
    int b, y, xs; strip_decode32(wave, b, y, xs);
    int x0 = xs*32;
    int xA = x0 + m;                     // A-tile0 pixel column; tile1 = +16
    v8f accz[2][2] = {};                 // [a-tile][n-tile]
    v8f accr[2][2] = {};
    const _Float16* xb = xp + (size_t)b*XP_H*XP_W*CPAD;
    for (int t = 0; t < 9; ++t) {
        int ty = t/3, tx = t - ty*3;
        const _Float16* pa = xb + ((size_t)(y+ty)*XP_W + (xA+tx))*CPAD + hi*8;
#pragma unroll
        for (int cc = 0; cc < 6; ++cc) {
            v16h a0 = load_a16(pa + cc*32);
            v16h a1 = load_a16(pa + 16*CPAD + cc*32);
            int kc = t*6 + cc;
            const _Float16* pz = wz + (size_t)kc*1024 + lane*16;
            const _Float16* pr = wr + (size_t)kc*1024 + lane*16;
            v16h b0 = *(const v16h*)(pz);
            v16h b1 = *(const v16h*)(pz + 512);
            v16h b2 = *(const v16h*)(pr);
            v16h b3 = *(const v16h*)(pr + 512);
            accz[0][0] = wmma16(a0, b0, accz[0][0]);
            accz[1][0] = wmma16(a1, b0, accz[1][0]);
            accz[0][1] = wmma16(a0, b1, accz[0][1]);
            accz[1][1] = wmma16(a1, b1, accz[1][1]);
            accr[0][0] = wmma16(a0, b2, accr[0][0]);
            accr[1][0] = wmma16(a1, b2, accr[1][0]);
            accr[0][1] = wmma16(a0, b3, accr[0][1]);
            accr[1][1] = wmma16(a1, b3, accr[1][1]);
        }
    }
    float bz0 = bz[m], bz1 = bz[m+16], br0 = br[m], br1 = br[m+16];
#pragma unroll
    for (int at = 0; at < 2; ++at) {
#pragma unroll
        for (int v = 0; v < 8; ++v) {
            int p = v + hi*8, xx = x0 + at*16 + p;
            size_t i0 = (((size_t)(b*HD + m))*HH + y)*WW + xx;
            size_t i1 = i0 + (size_t)16*HH*WW;
            zout[i0] = sigm(accz[at][0][v] + bz0);
            zout[i1] = sigm(accz[at][1][v] + bz1);
            rout[i0] = sigm(accr[at][0][v] + br0);
            rout[i1] = sigm(accr[at][1][v] + br1);
        }
    }
}

// ---------------- ConvGRU q + h update + h pack, M=32 ----------------
__global__ __launch_bounds__(256) void gru_q_kernel(
    const _Float16* __restrict__ xp, const _Float16* __restrict__ wq,
    const float* __restrict__ bq, const float* __restrict__ zbuf,
    const float* __restrict__ hidden, float* __restrict__ hout,
    _Float16* __restrict__ hp) {
    int wave = (int)((blockIdx.x*blockDim.x + threadIdx.x) >> 5);
    if (wave >= NWAVE32) return;
    int lane = threadIdx.x & 31, m = lane & 15, hi = lane >> 4;
    int b, y, xs; strip_decode32(wave, b, y, xs);
    int x0 = xs*32;
    int xA = x0 + m;
    v8f acc[2][2] = {};
    const _Float16* xb = xp + (size_t)b*XP_H*XP_W*CPAD;
    for (int t = 0; t < 9; ++t) {
        int ty = t/3, tx = t - ty*3;
        const _Float16* pa = xb + ((size_t)(y+ty)*XP_W + (xA+tx))*CPAD + hi*8;
#pragma unroll
        for (int cc = 0; cc < 6; ++cc) {
            v16h a0 = load_a16(pa + cc*32);
            v16h a1 = load_a16(pa + 16*CPAD + cc*32);
            const _Float16* pq = wq + (size_t)(t*6 + cc)*1024 + lane*16;
            v16h b0 = *(const v16h*)(pq);
            v16h b1 = *(const v16h*)(pq + 512);
            acc[0][0] = wmma16(a0, b0, acc[0][0]);
            acc[1][0] = wmma16(a1, b0, acc[1][0]);
            acc[0][1] = wmma16(a0, b1, acc[0][1]);
            acc[1][1] = wmma16(a1, b1, acc[1][1]);
        }
    }
    float bq0 = bq[m], bq1 = bq[m+16];
#pragma unroll
    for (int at = 0; at < 2; ++at) {
#pragma unroll
        for (int v = 0; v < 8; ++v) {
            int p = v + hi*8, xx = x0 + at*16 + p;
            size_t i0 = (((size_t)(b*HD + m))*HH + y)*WW + xx;
            size_t i1 = i0 + (size_t)16*HH*WW;
            float q0 = tanhf(acc[at][0][v] + bq0), q1 = tanhf(acc[at][1][v] + bq1);
            float z0 = zbuf[i0], z1 = zbuf[i1];
            float h0 = (1.f - z0)*hidden[i0] + z0*q0;
            float h1 = (1.f - z1)*hidden[i1] + z1*q1;
            hout[i0] = h0; hout[i1] = h1;
            size_t ho = (((size_t)b*HP_H + (y+2))*HP_W + (xx+2))*HD;
            hp[ho + m] = (_Float16)h0; hp[ho + m + 16] = (_Float16)h1;
        }
    }
}

// ---------------- dh1 & ch1 (3x3 dil2) share A-tiles, M=16 ----------------
__global__ __launch_bounds__(256) void heads33_kernel(
    const _Float16* __restrict__ hp, const _Float16* __restrict__ wd,
    const _Float16* __restrict__ wc, _Float16* __restrict__ t1,
    _Float16* __restrict__ c1) {
    int wave = (int)((blockIdx.x*blockDim.x + threadIdx.x) >> 5);
    if (wave >= NWAVE16) return;
    int lane = threadIdx.x & 31, m = lane & 15, hi = lane >> 4;
    int b, y, xs; strip_decode16(wave, b, y, xs);
    int x = xs*16 + m;
    v8f ad0 = {}, ad1 = {}, ac0 = {}, ac1 = {};
    const _Float16* hb = hp + (size_t)b*HP_H*HP_W*HD;
    for (int t = 0; t < 9; ++t) {
        int ty = t/3, tx = t - ty*3;
        v16h a = load_a16(hb + ((size_t)(y + 2*ty)*HP_W + (x + 2*tx))*HD + hi*8);
        const _Float16* pd = wd + (size_t)t*1024 + lane*16;
        const _Float16* pc = wc + (size_t)t*1024 + lane*16;
        ad0 = wmma16(a, *(const v16h*)(pd),       ad0);
        ad1 = wmma16(a, *(const v16h*)(pd + 512), ad1);
        ac0 = wmma16(a, *(const v16h*)(pc),       ac0);
        ac1 = wmma16(a, *(const v16h*)(pc + 512), ac1);
    }
#pragma unroll
    for (int v = 0; v < 8; ++v) {
        int p = v + hi*8, xx = xs*16 + p;
        size_t pix = ((size_t)(b*HH) + y)*WW + xx;
        _Float16* tp = t1 + pix*32; _Float16* cp = c1 + pix*32;
        tp[m]      = (_Float16)fmaxf(ad0[v], 0.f);
        tp[m + 16] = (_Float16)fmaxf(ad1[v], 0.f);
        cp[m]      = (_Float16)fmaxf(ac0[v], 0.f);
        cp[m + 16] = (_Float16)fmaxf(ac1[v], 0.f);
    }
}

// ---------------- dh2 1x1 32->64 + relu, M=32 ----------------
__global__ __launch_bounds__(256) void dh2_kernel(
    const _Float16* __restrict__ t1, const _Float16* __restrict__ w2,
    _Float16* __restrict__ t2) {
    int wave = (int)((blockIdx.x*blockDim.x + threadIdx.x) >> 5);
    if (wave >= NWAVE32) return;
    int lane = threadIdx.x & 31, m = lane & 15, hi = lane >> 4;
    int b, y, xs; strip_decode32(wave, b, y, xs);
    int x0 = xs*32;
    const _Float16* pa = t1 + (((size_t)(b*HH) + y)*WW + x0 + m)*32 + hi*8;
    v16h a0 = load_a16(pa);
    v16h a1 = load_a16(pa + 16*32);
    v8f acc[2][4] = {};
#pragma unroll
    for (int nt = 0; nt < 4; ++nt) {
        v16h bf = *(const v16h*)(w2 + nt*512 + lane*16);
        acc[0][nt] = wmma16(a0, bf, acc[0][nt]);
        acc[1][nt] = wmma16(a1, bf, acc[1][nt]);
    }
#pragma unroll
    for (int at = 0; at < 2; ++at) {
#pragma unroll
        for (int v = 0; v < 8; ++v) {
            int p = v + hi*8, xx = x0 + at*16 + p;
            _Float16* tp = t2 + (((size_t)(b*HH) + y)*WW + xx)*64;
#pragma unroll
            for (int nt = 0; nt < 4; ++nt)
                tp[nt*16 + m] = (_Float16)fmaxf(acc[at][nt][v], 0.f);
        }
    }
}

// ---------------- confidence 1x1 32->1 + sigmoid ----------------
__global__ void conf_kernel(const _Float16* __restrict__ c1, const float* __restrict__ ch2,
                            const float* __restrict__ cb2, float* __restrict__ c0out,
                            float* __restrict__ confout) {
    int i = blockIdx.x*blockDim.x + threadIdx.x;
    if (i >= (int)PIX) return;
    const _Float16* p = c1 + (size_t)i*32;
    float s = cb2[0];
#pragma unroll
    for (int c = 0; c < 32; ++c) s += (float)p[c]*ch2[c];
    c0out[i] = s;
    confout[i] = sigm(s);
}

// ---------------- fused dh3 (64->256) + softmax + windowed regression ----------------
__global__ __launch_bounds__(256) void dh3_softmax_kernel(
    const _Float16* __restrict__ t2, const _Float16* __restrict__ w3,
    const float* __restrict__ db3, float* __restrict__ prob, float* __restrict__ nd) {
    extern __shared__ float smem[];
    int lane = threadIdx.x & 31, m = lane & 15, hi = lane >> 4;
    float* lds = smem + (size_t)(threadIdx.x >> 5)*(NS*16);
    int wave = (int)((blockIdx.x*blockDim.x + threadIdx.x) >> 5);
    if (wave >= NWAVE16) return;
    int b, y, xs; strip_decode16(wave, b, y, xs);
    int x = xs*16 + m;
    const _Float16* pa = t2 + (((size_t)(b*HH) + y)*WW + x)*64 + hi*8;
    v8f acc[16] = {};
    for (int kc = 0; kc < 2; ++kc) {
        v16h a = load_a16(pa + kc*32);
        const _Float16* wb = w3 + (size_t)kc*(16*512) + lane*16;
#pragma unroll
        for (int nt = 0; nt < 16; ++nt)
            acc[nt] = wmma16(a, *(const v16h*)(wb + nt*512), acc[nt]);
    }
#pragma unroll
    for (int nt = 0; nt < 16; ++nt) {
        float bb = db3[nt*16 + m];
#pragma unroll
        for (int v = 0; v < 8; ++v) acc[nt][v] += bb;
    }
    // per-pixel softmax + argmax-window regression (pixel p lives in one 16-lane half-group)
    for (int v = 0; v < 8; ++v) {
        int p = v + hi*8;
        float lmax = acc[0][v];
#pragma unroll
        for (int nt = 1; nt < 16; ++nt) lmax = fmaxf(lmax, acc[nt][v]);
#pragma unroll
        for (int d = 1; d < 16; d <<= 1) lmax = fmaxf(lmax, __shfl_xor(lmax, d, 32));
        float pv[16], s = 0.f;
#pragma unroll
        for (int nt = 0; nt < 16; ++nt) { float e = __expf(acc[nt][v] - lmax); pv[nt] = e; s += e; }
#pragma unroll
        for (int d = 1; d < 16; d <<= 1) s += __shfl_xor(s, d, 32);
        float inv = 1.f/s;
        // argmax (first-max tie-break like jnp.argmax)
        float bvv = acc[0][v]; int bidx = m;
#pragma unroll
        for (int nt = 1; nt < 16; ++nt)
            if (acc[nt][v] > bvv) { bvv = acc[nt][v]; bidx = nt*16 + m; }
#pragma unroll
        for (int d = 1; d < 16; d <<= 1) {
            float ov = __shfl_xor(bvv, d, 32);
            int   oi = __shfl_xor(bidx, d, 32);
            if (ov > bvv || (ov == bvv && oi < bidx)) { bvv = ov; bidx = oi; }
        }
        // clipped window [-4,4] -> per-channel multiplicity counts (no cross-lane gather)
        float num = 0.f, den = 0.f;
#pragma unroll
        for (int nt = 0; nt < 16; ++nt) {
            int c = nt*16 + m;
            float cnt;
            if (c == 0)          { int t0 = 5 - bidx;   cnt = (float)(t0 > 0 ? t0 : 0); }
            else if (c == NS-1)  { int t0 = bidx - 250; cnt = (float)(t0 > 0 ? t0 : 0); }
            else { int d0 = c - bidx; if (d0 < 0) d0 = -d0; cnt = (d0 <= 4) ? 1.f : 0.f; }
            float pp = pv[nt]*inv;
            num += cnt*(float)c*pp;
            den += cnt*pp;
            lds[c*16 + p] = pp;            // stage prob for coalesced store
        }
#pragma unroll
        for (int d = 1; d < 16; d <<= 1) { num += __shfl_xor(num, d, 32); den += __shfl_xor(den, d, 32); }
        if (m == 0) {
            float reg = num/(1e-6f + den);
            nd[((size_t)b*HH + y)*WW + xs*16 + p] = reg*(1.f/(float)(NS - 1));
        }
    }
    __syncthreads();
    // coalesced prob writes: 16 consecutive x per half-group per channel
    for (int it = 0; it < 128; ++it) {
        int ch = it*2 + hi;
        prob[(((size_t)(b*NS + ch))*HH + y)*WW + xs*16 + m] = lds[ch*16 + m];
    }
}

// ---------------- host launcher ----------------
extern "C" void kernel_launch(void* const* d_in, const int* in_sizes, int n_in,
                              void* d_out, int out_size, void* d_ws, size_t ws_size,
                              hipStream_t stream) {
    (void)in_sizes; (void)n_in; (void)out_size; (void)ws_size;
    const float* hidden = (const float*)d_in[0];
    const float* depth  = (const float*)d_in[1];
    const float* corr   = (const float*)d_in[2];
    const float* Wz  = (const float*)d_in[3];  const float* bz  = (const float*)d_in[4];
    const float* Wr  = (const float*)d_in[5];  const float* br  = (const float*)d_in[6];
    const float* Wq  = (const float*)d_in[7];  const float* bq  = (const float*)d_in[8];
    const float* dh1 = (const float*)d_in[9];  const float* dh2 = (const float*)d_in[10];
    const float* dh3 = (const float*)d_in[11]; const float* db3 = (const float*)d_in[12];
    const float* ch1 = (const float*)d_in[13]; const float* ch2 = (const float*)d_in[14];
    const float* cb2 = (const float*)d_in[15];

    char* ws = (char*)d_ws;
    _Float16* xpack = (_Float16*)(ws + OFF_XPACK);
    _Float16* hpack = (_Float16*)(ws + OFF_HPACK);
    float*    zbuf  = (float*)(ws + OFF_Z);
    float*    rbuf  = (float*)(ws + OFF_R);
    _Float16* t1    = (_Float16*)(ws + OFF_T1);
    _Float16* c1    = (_Float16*)(ws + OFF_C1);
    _Float16* t2    = (_Float16*)(ws + OFF_T2);
    _Float16* wzp   = (_Float16*)(ws + OFF_WZ);
    _Float16* wrp   = (_Float16*)(ws + OFF_WR);
    _Float16* wqp   = (_Float16*)(ws + OFF_WQ);
    _Float16* wd1p  = (_Float16*)(ws + OFF_WD1);
    _Float16* wc1p  = (_Float16*)(ws + OFF_WC1);
    _Float16* wd2p  = (_Float16*)(ws + OFF_WD2);
    _Float16* wd3p  = (_Float16*)(ws + OFF_WD3);

    float* out      = (float*)d_out;
    float* h_out    = out;                                  // [4,32,256,320]
    float* nd_out   = out + (size_t)BN*HD*HH*WW;            // [4,1,256,320]
    float* prob_out = nd_out + PIX;                         // [4,256,256,320]
    float* conf_out = prob_out + (size_t)BN*NS*HH*WW;       // [4,1,256,320]
    float* c0_out   = conf_out + PIX;                       // [4,1,256,320]

    // weight packing (tiny)
    pack_w33_kernel<<<216, 256, 0, stream>>>(Wz,  wzp,  32, 161, 6);
    pack_w33_kernel<<<216, 256, 0, stream>>>(Wr,  wrp,  32, 161, 6);
    pack_w33_kernel<<<216, 256, 0, stream>>>(Wq,  wqp,  32, 161, 6);
    pack_w33_kernel<<<36,  256, 0, stream>>>(dh1, wd1p, 32, 32, 1);
    pack_w33_kernel<<<36,  256, 0, stream>>>(ch1, wc1p, 32, 32, 1);
    pack_w11_kernel<<<8,   256, 0, stream>>>(dh2, wd2p, 64, 32);
    pack_w11_kernel<<<64,  256, 0, stream>>>(dh3, wd3p, 256, 64);

    // activation packing
    {
        size_t n = XPACK_ELEMS;
        pack_x_kernel<<<(int)((n + 255)/256), 256, 0, stream>>>(hidden, depth, corr, xpack);
    }
    {
        size_t n = SZ_HPACK/4;
        zero32_kernel<<<(int)((n + 255)/256), 256, 0, stream>>>((uint32_t*)hpack, n);
    }

    // ConvGRU
    gru_zr_kernel<<<NBLK32, 256, 0, stream>>>(xpack, wzp, wrp, bz, br, zbuf, rbuf);
    {
        size_t n = PIX*HD;
        rhpack_kernel<<<(int)((n + 255)/256), 256, 0, stream>>>(rbuf, hidden, xpack);
    }
    gru_q_kernel<<<NBLK32, 256, 0, stream>>>(xpack, wqp, bq, zbuf, hidden, h_out, hpack);

    // heads
    heads33_kernel<<<NBLK16, 256, 0, stream>>>(hpack, wd1p, wc1p, t1, c1);
    dh2_kernel<<<NBLK32, 256, 0, stream>>>(t1, wd2p, t2);
    conf_kernel<<<(int)((PIX + 255)/256), 256, 0, stream>>>(c1, ch2, cb2, c0_out, conf_out);
    dh3_softmax_kernel<<<NBLK16, 256, 8*NS*16*sizeof(float), stream>>>(t2, wd3p, db3, prob_out, nd_out);
}
